// AF_LSTM_87488483819867
// MI455X (gfx1250) — matrix operand order repaired and checked by reading, not compiled, not run on hardware
//
#include <hip/hip_runtime.h>
#include <hip/hip_bf16.h>
#include <cstdint>
#include <cstddef>

// Problem sizes (fixed by the reference)
#define BB 64      // batch
#define SS 1024    // sequence
#define DD 128     // embed / hidden
#define GG 512     // 4*D gate width

typedef __attribute__((ext_vector_type(16))) __bf16 v16bf;
typedef __attribute__((ext_vector_type(8)))  __bf16 v8bf;
typedef __attribute__((ext_vector_type(8)))  float  v8f;

// ---- WMMA operand loaders (layouts per CDNA5 ISA §7.12.2) ----
// A (16x32 bf16): lane half 0 holds K 0..7 & 16..23, half 1 holds K 8..15 & 24..31
__device__ __forceinline__ v16bf load_a16(const __bf16* row, int k0) {
  v8bf lo = *(const v8bf*)(row + k0);
  v8bf hi = *(const v8bf*)(row + k0 + 16);
  v16bf r;
#pragma unroll
  for (int i = 0; i < 8; ++i) { r[i] = lo[i]; r[i + 8] = hi[i]; }
  return r;
}
// B (32x16 bf16), B = Wt so column n of B == row n of W: lane half h holds K 8h..8h+7 & 16+8h..16+8h+7
__device__ __forceinline__ v16bf load_b16(const __bf16* row, int kc, int half) {
  v8bf lo = *(const v8bf*)(row + kc * 32 + half * 8);
  v8bf hi = *(const v8bf*)(row + kc * 32 + half * 8 + 16);
  v16bf r;
#pragma unroll
  for (int i = 0; i < 8; ++i) { r[i] = lo[i]; r[i + 8] = hi[i]; }
  return r;
}
__device__ __forceinline__ v8f wmma_bf16(v16bf a, v16bf b, v8f c) {
  return __builtin_amdgcn_wmma_f32_16x16x32_bf16(false, a, false, b, (short)0, c, false, false);
}
__device__ __forceinline__ float sigmoidf_(float x) { return 1.0f / (1.0f + __expf(-x)); }

// ---------------------------------------------------------------------------
// 1) Embedding gather -> X (S,B,D) bf16
__global__ void k_embed(const int* __restrict__ new_list, const float* __restrict__ emb,
                        __bf16* __restrict__ Xbf) {
  int idx = blockIdx.x * 256 + threadIdx.x;           // flat over S*B*D
  int s = idx / (BB * DD);
  int rem = idx - s * (BB * DD);
  int b = rem / DD, d = rem - b * DD;
  int tok = new_list[b * SS + s];
  Xbf[idx] = (__bf16)emb[(size_t)tok * DD + d];
}

// 2) Convert weight matrices to bf16
__global__ void k_cvtW(const float* __restrict__ Wih, const float* __restrict__ Whh,
                       const float* __restrict__ Watt,
                       __bf16* __restrict__ WihB, __bf16* __restrict__ WhhB,
                       __bf16* __restrict__ WattB) {
  int idx = blockIdx.x * 256 + threadIdx.x;           // 65536 threads
  WihB[idx] = (__bf16)Wih[idx];
  WhhB[idx] = (__bf16)Whh[idx];
  if (idx < DD * DD) WattB[idx] = (__bf16)Watt[idx];
}

// 3) Precompute G = X @ W_ih^T + b_ih + b_hh   (M=65536, N=512, K=128) via WMMA
__global__ void k_input_gemm(const __bf16* __restrict__ Xbf, const __bf16* __restrict__ WihB,
                             const float* __restrict__ b_ih, const float* __restrict__ b_hh,
                             float* __restrict__ G) {
  int wave = (blockIdx.x << 3) + (threadIdx.x >> 5);  // 8 waves/block, 1 tile/wave
  int lane = threadIdx.x & 31, half = lane >> 4, lm = lane & 15;
  int mt = wave >> 5;                                  // 0..4095  (M tiles)
  int nt = wave & 31;                                  // 0..31    (N tiles)
  const __bf16* Arow = Xbf + (size_t)(mt * 16 + lm) * DD;
  int n = nt * 16 + lm;
  const __bf16* Brow = WihB + (size_t)n * DD;
  v8f acc = {};
#pragma unroll
  for (int kc = 0; kc < 4; ++kc) {
    v16bf a = load_a16(Arow, kc * 32 + half * 8);
    v16bf b = load_b16(Brow, kc, half);
    acc = wmma_bf16(a, b, acc);
  }
  float bias = b_ih[n] + b_hh[n];
#pragma unroll
  for (int r = 0; r < 8; ++r) {
    int row = mt * 16 + r + 8 * half;                  // C layout: M = r + 8*(lane/16)
    G[(size_t)row * GG + n] = acc[r] + bias;
  }
}

// 4) Persistent batch-parallel LSTM recurrence. 4 WGs, each owns 16 batch rows for all S steps.
//    W_hh is staged in LDS (131 KB) so the latency-critical per-step GEMM reads both operands
//    from LDS: total static LDS = 172 KB < 320 KB/WGP (one WG per WGP by construction).
__global__ void __launch_bounds__(512)
k_lstm(const float* __restrict__ G, const __bf16* __restrict__ WhhB, float* __restrict__ hs) {
  __shared__ __bf16 whh_s[GG * DD];  // 128 KB (B operand, resident all S steps)
  __shared__ __bf16 h_bf[16 * DD];   // 4 KB   (WMMA A operand for next step)
  __shared__ float  c_s[16 * DD];    // 8 KB
  __shared__ float  gates[16 * GG];  // 32 KB
  int tid = threadIdx.x, wg = blockIdx.x;
  // stage W_hh (bf16) into LDS: 8192 x uint4
  {
    const uint4* src = (const uint4*)WhhB;
    uint4* dst = (uint4*)whh_s;
    for (int t = tid; t < (GG * DD * 2) / 16; t += 512) dst[t] = src[t];
  }
  for (int t = tid; t < 16 * DD; t += 512) { h_bf[t] = (__bf16)0.0f; c_s[t] = 0.0f; }
  __syncthreads();
  int wave = tid >> 5, lane = tid & 31, half = lane >> 4, lm = lane & 15;
  for (int s = 0; s < SS; ++s) {
    { // prefetch next step's precomputed gates (global_prefetch_b8)
      int sn = (s + 1 < SS) ? s + 1 : s;
      const float* pf = G + ((size_t)sn * BB + wg * 16) * GG;
      __builtin_prefetch(pf + (size_t)tid * 16, 0, 0);
    }
#pragma unroll
    for (int ntl = 0; ntl < 2; ++ntl) {                // 16 waves x 2 tiles = 32 N-tiles
      int n = (wave * 2 + ntl) * 16 + lm;
      const __bf16* Brow = whh_s + (size_t)n * DD;
      const __bf16* Arow = h_bf + lm * DD;
      v8f acc = {};
#pragma unroll
      for (int kc = 0; kc < 4; ++kc) {
        v16bf a = load_a16(Arow, kc * 32 + half * 8);
        v16bf b = load_b16(Brow, kc, half);
        acc = wmma_bf16(a, b, acc);
      }
#pragma unroll
      for (int r = 0; r < 8; ++r) {
        int row = r + 8 * half;
        float gx = G[((size_t)s * BB + wg * 16 + row) * GG + n];
        gates[row * GG + n] = acc[r] + gx;
      }
    }
    __syncthreads();
    for (int t = tid; t < 16 * DD; t += 512) {
      int row = t >> 7, hh = t & 127;
      const float* gr = gates + row * GG;
      float gi = gr[hh], gf = gr[DD + hh], gg = gr[2 * DD + hh], go = gr[3 * DD + hh];
      float cc = sigmoidf_(gf) * c_s[t] + sigmoidf_(gi) * tanhf(gg);
      float hv = sigmoidf_(go) * tanhf(cc);
      c_s[t] = cc;
      h_bf[t] = (__bf16)hv;
      hs[((size_t)s * BB + wg * 16 + row) * DD + hh] = hv;   // raw h (h_L needs it)
    }
    __syncthreads();
  }
}

// 5) LayerNorm of hidden states, transposed to (B,S,D) bf16
__global__ void k_ln_hidden(const float* __restrict__ hs, const float* __restrict__ g,
                            const float* __restrict__ bta, __bf16* __restrict__ outB) {
  __shared__ float red[128];
  int rid = blockIdx.x;                 // s*B + b
  int s = rid / BB, b = rid - s * BB;
  int d = threadIdx.x;
  float x = hs[(size_t)rid * DD + d];
  red[d] = x; __syncthreads();
  for (int o = 64; o > 0; o >>= 1) { if (d < o) red[d] += red[d + o]; __syncthreads(); }
  float mean = red[0] * (1.0f / DD); __syncthreads();
  float dx = x - mean;
  red[d] = dx * dx; __syncthreads();
  for (int o = 64; o > 0; o >>= 1) { if (d < o) red[d] += red[d + o]; __syncthreads(); }
  float var = red[0] * (1.0f / DD);
  float y = dx * rsqrtf(var + 1e-5f) * g[d] + bta[d];
  outB[((size_t)b * SS + s) * DD + d] = (__bf16)y;
}

// 6) LayerNorm of aspect embedding -> f32 (B,D)
__global__ void k_ln_aspect(const int* __restrict__ aspect, const float* __restrict__ emb,
                            const float* __restrict__ g, const float* __restrict__ bta,
                            float* __restrict__ out) {
  __shared__ float red[128];
  int b = blockIdx.x, d = threadIdx.x;
  float x = emb[(size_t)aspect[b] * DD + d];
  red[d] = x; __syncthreads();
  for (int o = 64; o > 0; o >>= 1) { if (d < o) red[d] += red[d + o]; __syncthreads(); }
  float mean = red[0] * (1.0f / DD); __syncthreads();
  float dx = x - mean;
  red[d] = dx * dx; __syncthreads();
  for (int o = 64; o > 0; o >>= 1) { if (d < o) red[d] += red[d + o]; __syncthreads(); }
  float var = red[0] * (1.0f / DD);
  out[b * DD + d] = dx * rsqrtf(var + 1e-5f) * g[d] + bta[d];
}

// 7) Circular convolution as circulant GEMM: M[b,s,d] = sum_j hid[b,s,j]*a[b,(d-j)&127]
__global__ void k_circconv(const __bf16* __restrict__ hidB, const float* __restrict__ aln,
                           __bf16* __restrict__ Mout) {
  __shared__ __bf16 circT[DD * DD];     // 32 KB: circT[d*128 + k] = a[(d-k)&127]
  int b = blockIdx.x >> 3, schunk = blockIdx.x & 7;
  int tid = threadIdx.x;
  for (int t = tid; t < DD * DD; t += 256) {
    int d = t >> 7, k = t & 127;
    circT[t] = (__bf16)aln[b * DD + ((d - k) & 127)];
  }
  __syncthreads();
  int wave = tid >> 5, lane = tid & 31, half = lane >> 4, lm = lane & 15;
  int srow = schunk * 128 + wave * 16;
  const __bf16* Arow = hidB + ((size_t)b * SS + srow + lm) * DD;
#pragma unroll
  for (int nt = 0; nt < 8; ++nt) {
    const __bf16* Brow = circT + (nt * 16 + lm) * DD;
    v8f acc = {};
#pragma unroll
    for (int kc = 0; kc < 4; ++kc) {
      v16bf a = load_a16(Arow, kc * 32 + half * 8);
      v16bf bb = load_b16(Brow, kc, half);
      acc = wmma_bf16(a, bb, acc);
    }
#pragma unroll
    for (int r = 0; r < 8; ++r) {
      int row = srow + r + 8 * half;
      Mout[((size_t)b * SS + row) * DD + nt * 16 + lm] = (__bf16)acc[r];
    }
  }
}

// 8) scores[b,s] = sum_d mask[b,s]*tanh((M@W_att^T)[b,s,d]) * W_aw[d]   (fused WMMA + reduce)
__global__ void k_qk_score(const __bf16* __restrict__ Mbf, const __bf16* __restrict__ WattB,
                           const float* __restrict__ Waw, const float* __restrict__ mask,
                           float* __restrict__ scores) {
  __shared__ float sred[8][16][16];
  int b = blockIdx.x >> 3, schunk = blockIdx.x & 7;
  int tid = threadIdx.x;
  int wave = tid >> 5, lane = tid & 31, half = lane >> 4, lm = lane & 15;
  int st = schunk * 128 + wave * 16;
  const __bf16* Arow = Mbf + ((size_t)b * SS + st + lm) * DD;
  float p[8];
#pragma unroll
  for (int r = 0; r < 8; ++r) p[r] = 0.0f;
#pragma unroll
  for (int nt = 0; nt < 8; ++nt) {
    int n = nt * 16 + lm;
    const __bf16* Brow = WattB + (size_t)n * DD;
    v8f acc = {};
#pragma unroll
    for (int kc = 0; kc < 4; ++kc) {
      v16bf a = load_a16(Arow, kc * 32 + half * 8);
      v16bf bb = load_b16(Brow, kc, half);
      acc = wmma_bf16(a, bb, acc);
    }
    float wn = Waw[n];
#pragma unroll
    for (int r = 0; r < 8; ++r) {
      int row = st + r + 8 * half;
      p[r] += tanhf(acc[r]) * mask[b * SS + row] * wn;
    }
  }
#pragma unroll
  for (int r = 0; r < 8; ++r) sred[wave][r + 8 * half][lm] = p[r];
  __syncthreads();
  if (tid < 128) {
    int w = tid >> 4, row = tid & 15;
    float s = 0.0f;
#pragma unroll
    for (int i = 0; i < 16; ++i) s += sred[w][row][i];
    scores[b * SS + schunk * 128 + w * 16 + row] = s;
  }
}

// 9) softmax over S and r_raw[b,d] = sum_s weight*hid[b,s,d]
__global__ void k_softmax_r(const float* __restrict__ scores, const __bf16* __restrict__ hidB,
                            float* __restrict__ r_raw) {
  __shared__ float wbuf[SS];
  __shared__ float red[256];
  int b = blockIdx.x, tid = threadIdx.x;
  float lmax = -1e30f;
  for (int i = tid; i < SS; i += 256) { float v = scores[b * SS + i]; wbuf[i] = v; lmax = fmaxf(lmax, v); }
  red[tid] = lmax; __syncthreads();
  for (int o = 128; o > 0; o >>= 1) { if (tid < o) red[tid] = fmaxf(red[tid], red[tid + o]); __syncthreads(); }
  float m = red[0]; __syncthreads();
  float lsum = 0.0f;
  for (int i = tid; i < SS; i += 256) { float e = __expf(wbuf[i] - m); wbuf[i] = e; lsum += e; }
  red[tid] = lsum; __syncthreads();
  for (int o = 128; o > 0; o >>= 1) { if (tid < o) red[tid] += red[tid + o]; __syncthreads(); }
  float inv = 1.0f / red[0]; __syncthreads();
  int d = tid & 127, sh = tid >> 7;
  float acc = 0.0f;
  for (int s = sh * 512; s < sh * 512 + 512; ++s)
    acc += wbuf[s] * (float)hidB[((size_t)b * SS + s) * DD + d];
  red[tid] = acc; __syncthreads();
  if (tid < 128) r_raw[b * DD + tid] = (red[tid] + red[tid + 128]) * inv;
}

// 10) r = tanh(r@W_rout^T + hL@W_hout^T); result = softmax(r@W_cls^T + b_cls)
__global__ void k_final(const float* __restrict__ r_raw, const float* __restrict__ hs,
                        const float* __restrict__ Wrout, const float* __restrict__ Whout,
                        const float* __restrict__ Wcls, const float* __restrict__ bcls,
                        float* __restrict__ out) {
  __shared__ float rr[DD];
  int b = blockIdx.x, d = threadIdx.x;
  const float* rv = r_raw + b * DD;
  const float* hL = hs + ((size_t)(SS - 1) * BB + b) * DD;
  float a1 = 0.0f, a2 = 0.0f;
  for (int j = 0; j < DD; ++j) { a1 += rv[j] * Wrout[d * DD + j]; a2 += hL[j] * Whout[d * DD + j]; }
  float val = tanhf(a1 + a2);
  rr[d] = val;
  out[BB * 2 + b * DD + d] = val;        // r output, after result (B,2)
  __syncthreads();
  if (d == 0) {
    float l0 = bcls[0], l1 = bcls[1];
    for (int j = 0; j < DD; ++j) { l0 += rr[j] * Wcls[j]; l1 += rr[j] * Wcls[DD + j]; }
    float mm = fmaxf(l0, l1);
    float e0 = __expf(l0 - mm), e1 = __expf(l1 - mm), inv = 1.0f / (e0 + e1);
    out[b * 2 + 0] = e0 * inv;
    out[b * 2 + 1] = e1 * inv;
  }
}

// ---------------------------------------------------------------------------
extern "C" void kernel_launch(void* const* d_in, const int* in_sizes, int n_in,
                              void* d_out, int out_size, void* d_ws, size_t ws_size,
                              hipStream_t stream) {
  (void)in_sizes; (void)n_in; (void)out_size; (void)ws_size;
  const int*   new_list = (const int*)d_in[0];
  const float* new_mask = (const float*)d_in[1];
  const int*   aspect   = (const int*)d_in[2];
  const float* emb      = (const float*)d_in[4];
  const float* W_ih     = (const float*)d_in[5];
  const float* W_hh     = (const float*)d_in[6];
  const float* b_ih     = (const float*)d_in[7];
  const float* b_hh     = (const float*)d_in[8];
  const float* an_g     = (const float*)d_in[9];
  const float* an_b     = (const float*)d_in[10];
  const float* hn_g     = (const float*)d_in[11];
  const float* hn_b     = (const float*)d_in[12];
  const float* W_att    = (const float*)d_in[13];
  const float* W_aw     = (const float*)d_in[14];
  const float* W_rout   = (const float*)d_in[15];
  const float* W_hout   = (const float*)d_in[16];
  const float* W_cls    = (const float*)d_in[17];
  const float* b_cls    = (const float*)d_in[18];
  float* out = (float*)d_out;

  // Workspace carve-up (~220 MB, dominated by f32 precomputed gates)
  char* ws = (char*)d_ws;
  size_t off = 0;
  auto take = [&](size_t bytes) {
    char* p = ws + off;
    off = (off + bytes + 255) & ~(size_t)255;
    return p;
  };
  __bf16* Xbf    = (__bf16*)take((size_t)SS * BB * DD * 2);
  __bf16* WihB   = (__bf16*)take((size_t)GG * DD * 2);
  __bf16* WhhB   = (__bf16*)take((size_t)GG * DD * 2);
  __bf16* WattB  = (__bf16*)take((size_t)DD * DD * 2);
  float*  G      = (float*) take((size_t)SS * BB * GG * 4);
  float*  hs     = (float*) take((size_t)SS * BB * DD * 4);
  __bf16* hidB   = (__bf16*)take((size_t)BB * SS * DD * 2);
  float*  aln    = (float*) take((size_t)BB * DD * 4);
  __bf16* Mbf    = (__bf16*)take((size_t)BB * SS * DD * 2);
  float*  scores = (float*) take((size_t)BB * SS * 4);
  float*  r_raw  = (float*) take((size_t)BB * DD * 4);

  k_embed<<<(SS * BB * DD) / 256, 256, 0, stream>>>(new_list, emb, Xbf);
  k_cvtW<<<(GG * DD) / 256, 256, 0, stream>>>(W_ih, W_hh, W_att, WihB, WhhB, WattB);
  k_input_gemm<<<(SS * BB / 16) * (GG / 16) / 8, 256, 0, stream>>>(Xbf, WihB, b_ih, b_hh, G);
  k_lstm<<<BB / 16, 512, 0, stream>>>(G, WhhB, hs);
  k_ln_hidden<<<SS * BB, 128, 0, stream>>>(hs, hn_g, hn_b, hidB);
  k_ln_aspect<<<BB, 128, 0, stream>>>(aspect, emb, an_g, an_b, aln);
  k_circconv<<<BB * (SS / 128), 256, 0, stream>>>(hidB, aln, Mbf);
  k_qk_score<<<BB * (SS / 128), 256, 0, stream>>>(Mbf, WattB, W_aw, new_mask, scores);
  k_softmax_r<<<BB, 256, 0, stream>>>(scores, hidB, r_raw);
  k_final<<<BB, 128, 0, stream>>>(r_raw, hs, W_rout, W_hout, W_cls, b_cls, out);
}
